// SelfAttention_73598559584522
// MI455X (gfx1250) — matrix operand order repaired and checked by reading, not compiled, not run on hardware
//
#include <hip/hip_runtime.h>
#include <hip/hip_bf16.h>
#include <math.h>
#include <stdint.h>

// Problem constants (match reference)
#define B_    2
#define S_    2048
#define DIM_  4096
#define NH_   32
#define NKV_  8
#define HD_   128
#define NREP_ 4
#define KVD_  (NKV_ * HD_)   // 1024

typedef __bf16 bf16;
typedef __attribute__((ext_vector_type(16))) __bf16 v16bf;
typedef __attribute__((ext_vector_type(8)))  __bf16 v8bf;
typedef __attribute__((ext_vector_type(8)))  float  v8f;
typedef __attribute__((ext_vector_type(4)))  int    v4i;

#define AS1 __attribute__((address_space(1)))
#define AS3 __attribute__((address_space(3)))

// ---------------------------------------------------------------------------
// Async global -> LDS copy (GLOBAL_LOAD_ASYNC_TO_LDS_B128, ASYNCcnt-tracked).
// Per-lane: LDS[ldst] = MEM[gsrc] (16 bytes).
// Builtin signature (per compiler diagnostic): (v4i* global, v4i* lds, Ii, Ii)
// ---------------------------------------------------------------------------
static __device__ __forceinline__ void async_b128(const void* gsrc, void* ldst) {
  __builtin_amdgcn_global_load_async_to_lds_b128(
      (AS1 v4i*)(uintptr_t)gsrc,
      (AS3 v4i*)(uint32_t)(uintptr_t)ldst,
      0, 0);
}
static __device__ __forceinline__ void wait_async0() {
  asm volatile("s_wait_asynccnt 0x0" ::: "memory");
}

// ---------------------------------------------------------------------------
// WMMA fragment loaders (V_WMMA_F32_16X16X32_BF16 operand layouts, ISA 7.12.2)
// A-matrix 16x32: lanes 0-15 -> M=lane, K = {0..7, 16..23}; lanes 16-31 same M,
//                 K = {8..15, 24..31}.  Source: row-major, K contiguous.
// ---------------------------------------------------------------------------
static __device__ __forceinline__ v16bf load_frag_a(const bf16* base, int ld, int lane) {
  const int hf = (lane >> 4) & 1;
  const int r  = lane & 15;
  const bf16* p = base + (size_t)r * ld + hf * 8;
  v8bf lo = *(const v8bf*)(p);        // K = hf*8 + 0..7
  v8bf hi = *(const v8bf*)(p + 16);   // K = 16 + hf*8 + 0..7
  v16bf f;
#pragma unroll
  for (int i = 0; i < 8; ++i) { f[i] = lo[i]; f[i + 8] = hi[i]; }
  return f;
}

// B-matrix 32x16: lanes 0-15 -> N=lane, K = {0..7, 16..23}; lanes 16-31 -> K = {8..15, 24..31}.
// Source: N-major rows, K contiguous (i.e. W[N][K]).
static __device__ __forceinline__ v16bf load_frag_b(const bf16* base, int ld, int lane) {
  const int hf = (lane >> 4) & 1;
  const int n  = lane & 15;
  const bf16* p = base + (size_t)n * ld + hf * 8;
  v8bf lo = *(const v8bf*)(p);        // K = hf*8 + 0..7
  v8bf hi = *(const v8bf*)(p + 16);   // K = 16 + hf*8 + 0..7
  v16bf f;
#pragma unroll
  for (int i = 0; i < 8; ++i) { f[i] = lo[i]; f[i + 8] = hi[i]; }
  return f;
}

// ---------------------------------------------------------------------------
// fp32 -> bf16 conversion
// ---------------------------------------------------------------------------
__global__ void cvt_f32_bf16(const float* __restrict__ in, bf16* __restrict__ out, int n) {
  int i = (blockIdx.x * blockDim.x + threadIdx.x) * 4;
  if (i + 3 < n) {
    float4 v = *(const float4*)(in + i);
    out[i]     = (bf16)v.x;
    out[i + 1] = (bf16)v.y;
    out[i + 2] = (bf16)v.z;
    out[i + 3] = (bf16)v.w;
  } else {
    for (; i < n; ++i) out[i] = (bf16)in[i];
  }
}

// ---------------------------------------------------------------------------
// RoPE on bf16 tensor laid out (B, S, H, 128); pairs are consecutive elements.
// ---------------------------------------------------------------------------
__global__ void rope_kernel(bf16* __restrict__ t, const float* __restrict__ cosT,
                            const float* __restrict__ sinT, int H, int total) {
  int i = blockIdx.x * blockDim.x + threadIdx.x;
  if (i >= total) return;
  int p   = i & 63;          // pair index within head
  int bsh = i >> 6;          // flattened (b, s, h)
  int s   = (bsh / H) % S_;
  size_t base = (size_t)bsh * HD_ + (size_t)p * 2;
  float c  = cosT[s * 64 + p];
  float sn = sinT[s * 64 + p];
  float t0 = (float)t[base];
  float t1 = (float)t[base + 1];
  t[base]     = (bf16)(t0 * c - t1 * sn);
  t[base + 1] = (bf16)(t0 * sn + t1 * c);
}

// ---------------------------------------------------------------------------
// Tiled WMMA GEMM:  C[M,N] = A[M,K] * W[N,K]^T   (bf16 in, f32 accumulate)
// Block 256 thr = 8 waves, tile 128x128, BK=32. Wave computes 64x32.
// Double-buffered: async global->LDS copy of tile k+1 overlaps WMMA of tile k.
// ---------------------------------------------------------------------------
template <bool OUT_BF16>
__global__ __launch_bounds__(256) void gemm_bf16_wmma(const bf16* __restrict__ A,
                                                      const bf16* __restrict__ W,
                                                      void* __restrict__ Cout,
                                                      int M, int N, int K) {
  __shared__ __align__(16) bf16 lA[2][128 * 32];
  __shared__ __align__(16) bf16 lW[2][128 * 32];

  const int tid  = threadIdx.x;
  const int lane = tid & 31;
  const int wave = tid >> 5;
  const int wr   = wave >> 2;   // 0..1 -> M offset
  const int wc   = wave & 3;    // 0..3 -> N offset
  const int bm   = blockIdx.y * 128;
  const int bn   = blockIdx.x * 128;
  const int hf   = lane >> 4;
  const int lcol = lane & 15;

  v8f acc[4][2];
#pragma unroll
  for (int i = 0; i < 4; ++i)
#pragma unroll
    for (int j = 0; j < 2; ++j)
#pragma unroll
      for (int r = 0; r < 8; ++r) acc[i][j][r] = 0.f;

  const int lrow0 = tid >> 2;         // 0..63
  const int lk0   = (tid & 3) * 8;    // 0,8,16,24

  auto issue_tile = [&](int buf, int k0) {
    async_b128(A + (size_t)(bm + lrow0) * K + k0 + lk0,        &lA[buf][lrow0 * 32 + lk0]);
    async_b128(A + (size_t)(bm + lrow0 + 64) * K + k0 + lk0,   &lA[buf][(lrow0 + 64) * 32 + lk0]);
    async_b128(W + (size_t)(bn + lrow0) * K + k0 + lk0,        &lW[buf][lrow0 * 32 + lk0]);
    async_b128(W + (size_t)(bn + lrow0 + 64) * K + k0 + lk0,   &lW[buf][(lrow0 + 64) * 32 + lk0]);
  };

  const int nk = K / 32;
  issue_tile(0, 0);
  wait_async0();
  __syncthreads();

  for (int kt = 0; kt < nk; ++kt) {
    const int buf = kt & 1;
    if (kt + 1 < nk) issue_tile(buf ^ 1, (kt + 1) * 32);

    v16bf af[4], wf[2];
#pragma unroll
    for (int i = 0; i < 4; ++i)
      af[i] = load_frag_a(&lA[buf][(wr * 64 + i * 16) * 32], 32, lane);
#pragma unroll
    for (int j = 0; j < 2; ++j)
      wf[j] = load_frag_b(&lW[buf][(wc * 32 + j * 16) * 32], 32, lane);

#pragma unroll
    for (int i = 0; i < 4; ++i)
#pragma unroll
      for (int j = 0; j < 2; ++j)
        acc[i][j] = __builtin_amdgcn_wmma_f32_16x16x32_bf16(
            false, af[i], false, wf[j], (short)0, acc[i][j], false, false);

    wait_async0();     // next tile landed in the other buffer
    __syncthreads();   // all waves done reading current buffer
  }

  // C layout: VGPR r, lane -> row = r + 8*hf, col = lane&15
#pragma unroll
  for (int i = 0; i < 4; ++i) {
#pragma unroll
    for (int j = 0; j < 2; ++j) {
#pragma unroll
      for (int r = 0; r < 8; ++r) {
        int m = bm + wr * 64 + i * 16 + r + 8 * hf;
        int n = bn + wc * 32 + j * 16 + lcol;
        if (OUT_BF16)
          ((bf16*)Cout)[(size_t)m * N + n] = (bf16)acc[i][j][r];
        else
          ((float*)Cout)[(size_t)m * N + n] = acc[i][j][r];
      }
    }
  }
}

// ---------------------------------------------------------------------------
// Flash attention (causal, GQA). One WG = (b, h, 64 q-rows), 4 waves.
// Wave owns 16 q-rows; KV tiles of 64 staged in LDS (K via async copy).
// ---------------------------------------------------------------------------
__global__ __launch_bounds__(128) void attn_kernel(const bf16* __restrict__ qt,
                                                   const bf16* __restrict__ kt,
                                                   const bf16* __restrict__ vt,
                                                   bf16* __restrict__ ot) {
  const int qblk = blockIdx.x;          // 0..S/64-1
  const int h    = blockIdx.y;          // 0..31
  const int b    = blockIdx.z;          // 0..1
  const int kvh  = h / NREP_;
  const int tid  = threadIdx.x;
  const int wave = tid >> 5;
  const int lane = tid & 31;
  const int hf   = lane >> 4;
  const int lcol = lane & 15;

  __shared__ __align__(16) bf16 ldsK[64 * HD_];     // [kv][d]     16 KB
  __shared__ __align__(16) bf16 ldsV[HD_ * 64];     // [d][kv]     16 KB (transposed)
  __shared__ __align__(16) bf16 ldsP[4][16 * 64];   // per-wave P   8 KB

  const int q0   = qblk * 64;
  const int qrow = q0 + wave * 16;

  // Preload this wave's Q fragments (16 x 128 = 4 K-slices)
  const bf16* qbase = qt + ((size_t)(b * S_ + qrow)) * DIM_ + h * HD_;
  v16bf qf[4];
#pragma unroll
  for (int kk = 0; kk < 4; ++kk) qf[kk] = load_frag_a(qbase + kk * 32, DIM_, lane);

  v8f o[8];
#pragma unroll
  for (int c = 0; c < 8; ++c)
#pragma unroll
    for (int r = 0; r < 8; ++r) o[c][r] = 0.f;

  float mrow[8], lrow[8];
#pragma unroll
  for (int r = 0; r < 8; ++r) { mrow[r] = -__builtin_inff(); lrow[r] = 0.f; }

  const float scale = 0.08838834764831845f;  // 1/sqrt(128)

  for (int kbk = 0; kbk <= qblk; ++kbk) {
    const int kv0 = kbk * 64;
    const bf16* ksrc = kt + ((size_t)(b * S_ + kv0)) * KVD_ + kvh * HD_;
    const bf16* vsrc = vt + ((size_t)(b * S_ + kv0)) * KVD_ + kvh * HD_;

    // K tile -> LDS via async DMA (direct copy, [kv][d]): 8 x b128 per thread
    {
      int row = tid >> 1;
      int c0  = (tid & 1) * 64;
      const bf16* src = ksrc + (size_t)row * KVD_ + c0;
      bf16* dst = &ldsK[row * HD_ + c0];
#pragma unroll
      for (int i = 0; i < 8; ++i)
        async_b128(src + i * 8, dst + i * 8);
    }
    // V tile -> LDS transposed ([d][kv]) so P*V uses B-frag (K contiguous)
    {
      int kv = tid >> 1;
      int d0 = (tid & 1) * 64;
      const bf16* src = vsrc + (size_t)kv * KVD_ + d0;
#pragma unroll
      for (int i = 0; i < 8; ++i) {
        v8bf v = *(const v8bf*)(src + i * 8);
#pragma unroll
        for (int u = 0; u < 8; ++u) ldsV[(d0 + i * 8 + u) * 64 + kv] = v[u];
      }
    }
    wait_async0();
    __syncthreads();

    // S = Q * K^T  (16 x 64)
    v8f s4[4];
#pragma unroll
    for (int j = 0; j < 4; ++j) {
      v8f a;
#pragma unroll
      for (int r = 0; r < 8; ++r) a[r] = 0.f;
#pragma unroll
      for (int kk = 0; kk < 4; ++kk) {
        v16bf kf = load_frag_b(&ldsK[(j * 16) * HD_ + kk * 32], HD_, lane);
        a = __builtin_amdgcn_wmma_f32_16x16x32_bf16(
            false, qf[kk], false, kf, (short)0, a, false, false);
      }
      s4[j] = a;
    }

    // scale + causal mask (only diagonal block needs per-element mask)
    const bool diag = (kbk == qblk);
#pragma unroll
    for (int j = 0; j < 4; ++j) {
      int col = kv0 + j * 16 + lcol;
#pragma unroll
      for (int r = 0; r < 8; ++r) {
        float v = s4[j][r] * scale;
        int row = qrow + r + 8 * hf;
        if (diag && col > row) v = -__builtin_inff();
        s4[j][r] = v;
      }
    }

    // online softmax: row max across 4 tiles + 16 lanes of this half
    float alpha[8];
#pragma unroll
    for (int r = 0; r < 8; ++r) {
      float mx = s4[0][r];
#pragma unroll
      for (int j = 1; j < 4; ++j) mx = fmaxf(mx, s4[j][r]);
#pragma unroll
      for (int off = 1; off < 16; off <<= 1) mx = fmaxf(mx, __shfl_xor(mx, off, 32));
      float mo = mrow[r];
      float mn = fmaxf(mo, mx);
      mrow[r]  = mn;
      alpha[r] = __expf(mo - mn);   // 0 on first visit (mo = -inf)
    }

    // P = exp(S - m), row sums
    float ls[8];
#pragma unroll
    for (int r = 0; r < 8; ++r) ls[r] = 0.f;
#pragma unroll
    for (int j = 0; j < 4; ++j)
#pragma unroll
      for (int r = 0; r < 8; ++r) {
        float p = __expf(s4[j][r] - mrow[r]);
        s4[j][r] = p;
        ls[r] += p;
      }
#pragma unroll
    for (int r = 0; r < 8; ++r) {
#pragma unroll
      for (int off = 1; off < 16; off <<= 1) ls[r] += __shfl_xor(ls[r], off, 32);
      lrow[r] = lrow[r] * alpha[r] + ls[r];
    }

    // rescale O
#pragma unroll
    for (int c = 0; c < 8; ++c)
#pragma unroll
      for (int r = 0; r < 8; ++r) o[c][r] *= alpha[r];

    // stage P (C-layout -> row-major bf16) in per-wave LDS
    bf16* pst = &ldsP[wave][0];
#pragma unroll
    for (int j = 0; j < 4; ++j)
#pragma unroll
      for (int r = 0; r < 8; ++r)
        pst[(r + 8 * hf) * 64 + j * 16 + lcol] = (bf16)s4[j][r];

    // O += P * V  (KV dim 64 = 2 K-slices)
#pragma unroll
    for (int t2 = 0; t2 < 2; ++t2) {
      v16bf pf = load_frag_a(pst + t2 * 32, 64, lane);
#pragma unroll
      for (int c = 0; c < 8; ++c) {
        v16bf vf = load_frag_b(&ldsV[(c * 16) * 64 + t2 * 32], 64, lane);
        o[c] = __builtin_amdgcn_wmma_f32_16x16x32_bf16(
            false, pf, false, vf, (short)0, o[c], false, false);
      }
    }
    __syncthreads();
  }

  // epilogue: O / l -> bf16, layout (B,S,H,128)
  bf16* obase = ot + ((size_t)(b * S_ + qrow)) * DIM_ + h * HD_;
#pragma unroll
  for (int r = 0; r < 8; ++r) {
    float inv = 1.f / lrow[r];
    int row = r + 8 * hf;
#pragma unroll
    for (int c = 0; c < 8; ++c)
      obase[(size_t)row * DIM_ + c * 16 + lcol] = (bf16)(o[c][r] * inv);
  }
}

// ---------------------------------------------------------------------------
// Host-side orchestration
// ---------------------------------------------------------------------------
extern "C" void kernel_launch(void* const* d_in, const int* in_sizes, int n_in,
                              void* d_out, int out_size, void* d_ws, size_t ws_size,
                              hipStream_t stream) {
  (void)in_sizes; (void)n_in; (void)out_size; (void)ws_size;

  const float* x  = (const float*)d_in[0];
  const float* wq = (const float*)d_in[1];
  const float* wk = (const float*)d_in[2];
  const float* wv = (const float*)d_in[3];
  const float* wo = (const float*)d_in[4];
  const float* fc = (const float*)d_in[5];
  const float* fs = (const float*)d_in[6];

  const size_t MBS = (size_t)B_ * S_;   // 4096 rows

  char* ws = (char*)d_ws;
  size_t off = 0;
  auto alloc = [&](size_t bytes) {
    char* p = ws + off;
    off += (bytes + 255) & ~(size_t)255;
    return p;
  };
  bf16* xb  = (bf16*)alloc(MBS * DIM_ * sizeof(bf16));
  bf16* wqb = (bf16*)alloc((size_t)DIM_ * DIM_ * sizeof(bf16));
  bf16* wkb = (bf16*)alloc((size_t)KVD_ * DIM_ * sizeof(bf16));
  bf16* wvb = (bf16*)alloc((size_t)KVD_ * DIM_ * sizeof(bf16));
  bf16* wob = (bf16*)alloc((size_t)DIM_ * DIM_ * sizeof(bf16));
  bf16* qbf = (bf16*)alloc(MBS * DIM_ * sizeof(bf16));
  bf16* kbf = (bf16*)alloc(MBS * KVD_ * sizeof(bf16));
  bf16* vbf = (bf16*)alloc(MBS * KVD_ * sizeof(bf16));
  bf16* abf = (bf16*)alloc(MBS * DIM_ * sizeof(bf16));

  auto cvt = [&](const float* in, bf16* outp, size_t n) {
    int blocks = (int)((n / 4 + 255) / 256);
    cvt_f32_bf16<<<blocks, 256, 0, stream>>>(in, outp, (int)n);
  };
  cvt(x,  xb,  MBS * DIM_);
  cvt(wq, wqb, (size_t)DIM_ * DIM_);
  cvt(wk, wkb, (size_t)KVD_ * DIM_);
  cvt(wv, wvb, (size_t)KVD_ * DIM_);
  cvt(wo, wob, (size_t)DIM_ * DIM_);

  // QKV projections (bf16 out)
  gemm_bf16_wmma<true><<<dim3(DIM_ / 128, (int)(MBS / 128)), 256, 0, stream>>>(
      xb, wqb, qbf, (int)MBS, DIM_, DIM_);
  gemm_bf16_wmma<true><<<dim3(KVD_ / 128, (int)(MBS / 128)), 256, 0, stream>>>(
      xb, wkb, kbf, (int)MBS, KVD_, DIM_);
  gemm_bf16_wmma<true><<<dim3(KVD_ / 128, (int)(MBS / 128)), 256, 0, stream>>>(
      xb, wvb, vbf, (int)MBS, KVD_, DIM_);

  // RoPE on Q and K (in place)
  {
    int totq = B_ * S_ * NH_ * (HD_ / 2);
    rope_kernel<<<(totq + 255) / 256, 256, 0, stream>>>(qbf, fc, fs, NH_, totq);
    int totk = B_ * S_ * NKV_ * (HD_ / 2);
    rope_kernel<<<(totk + 255) / 256, 256, 0, stream>>>(kbf, fc, fs, NKV_, totk);
  }

  // Causal GQA flash attention
  attn_kernel<<<dim3(S_ / 64, NH_, B_), 128, 0, stream>>>(qbf, kbf, vbf, abf);

  // Output projection (fp32 out -> d_out)
  gemm_bf16_wmma<false><<<dim3(DIM_ / 128, (int)(MBS / 128)), 256, 0, stream>>>(
      abf, wob, d_out, (int)MBS, DIM_, DIM_);
}
